// encoder_stage_76209899700316
// MI455X (gfx1250) — matrix operand order repaired and checked by reading, not compiled, not run on hardware
//
#include <hip/hip_runtime.h>
#include <hip/hip_bf16.h>
#include <hip/hip_fp16.h>

typedef __attribute__((ext_vector_type(16))) _Float16 v16h;
typedef __attribute__((ext_vector_type(8)))  _Float16 v8h;
typedef __attribute__((ext_vector_type(8)))  float    v8f;

#define BN_EPS 1e-5f

__device__ __forceinline__ float gelu_exact(float x) {
    return 0.5f * x * (1.0f + erff(x * 0.70710678118654752440f));
}

// ---------------------------------------------------------------------------
// Kernel 1: gather new_xyz = xyz[fps_idx]  -> d_out[0 .. B*S*3)
// ---------------------------------------------------------------------------
__global__ void k_newxyz(const float* __restrict__ xyz, const long long* __restrict__ fps,
                         float* __restrict__ out, int B, int N, int S) {
    int i = blockIdx.x * blockDim.x + threadIdx.x;  // over B*S
    if (i >= B * S) return;
    int b = i / S;
    long long n = fps[i];
    const float* src = xyz + ((long long)b * N + n) * 3;
    float* dst = out + (long long)i * 3;
    dst[0] = src[0]; dst[1] = src[1]; dst[2] = src[2];
}

// ---------------------------------------------------------------------------
// Kernel 2: KNN (K=32 smallest sq-dist, lower-index tiebreak = lax.top_k) and
// feature build (position-major): feat[(q*32+k)*128 + c]
// ---------------------------------------------------------------------------
#define KNN_T 256
__global__ void k_knn_feat(const float* __restrict__ xyz, const float* __restrict__ x,
                           const long long* __restrict__ fps, _Float16* __restrict__ feat,
                           int B, int N, int S) {
    __shared__ float dist[4096];
    __shared__ float rv[KNN_T];
    __shared__ int   ri[KNN_T];
    __shared__ int   sel[32];
    __shared__ float cf[64];

    int q = blockIdx.x;            // b*S + s
    int b = q / S;
    int t = threadIdx.x;
    long long nc = fps[q];
    const float* xb = xyz + (long long)b * N * 3;
    float cx = xb[nc * 3 + 0], cy = xb[nc * 3 + 1], cz = xb[nc * 3 + 2];

    const int per = 4096 / KNN_T;  // 16, contiguous chunk per thread
    for (int i = 0; i < per; i++) {
        int n = t * per + i;
        float dx = xb[n * 3 + 0] - cx;
        float dy = xb[n * 3 + 1] - cy;
        float dz = xb[n * 3 + 2] - cz;
        dist[n] = dx * dx + dy * dy + dz * dz;
    }
    if (t < 64) cf[t] = x[((long long)b * 64 + t) * N + nc];
    __syncthreads();

    for (int k = 0; k < 32; k++) {
        float bv = 3.402823466e38f; int bi = 0x7fffffff;
        for (int i = 0; i < per; i++) {           // ascending scan: lowest idx on tie
            int n = t * per + i;
            float v = dist[n];
            if (v < bv) { bv = v; bi = n; }
        }
        rv[t] = bv; ri[t] = bi;
        __syncthreads();
        for (int off = KNN_T / 2; off > 0; off >>= 1) {
            if (t < off) {
                float v2 = rv[t + off]; int i2 = ri[t + off];
                if (v2 < rv[t] || (v2 == rv[t] && i2 < ri[t])) { rv[t] = v2; ri[t] = i2; }
            }
            __syncthreads();
        }
        if (t == 0) { sel[k] = ri[0]; dist[ri[0]] = 3.402823466e38f; }
        __syncthreads();
    }

    // position-major write: consecutive threads -> consecutive channels
    for (int e = t; e < 32 * 128; e += KNN_T) {
        int k = e >> 7, c = e & 127;
        int n = sel[k];
        float v = (c < 64) ? (x[((long long)b * 64 + c) * N + n] - cf[c]) : cf[c - 64];
        feat[((long long)q * 32 + k) * 128 + c] = (_Float16)v;
    }
}

// ---------------------------------------------------------------------------
// Kernel 3: GEMM  H[p,o] = sum_c W[o,c] * Y[p,c]   (position-major NHWC)
// v_wmma_f32_16x16x32_f16; 8 waves/WG, wave w owns output channels [16w,16w+16).
// Y tiles staged with double-buffered GLOBAL_LOAD_ASYNC_TO_LDS_B128 (ASYNCcnt),
// overlapping global->LDS DMA with WMMA of the current tile.
// ---------------------------------------------------------------------------
#define GEMM_T 256
#define WSTR 136   // padded W row stride (f16): 272B = 68 banks
#define YSTR 136   // padded Y-tile row stride (f16)
__global__ void k_gemm(const _Float16* __restrict__ Yact, const float* __restrict__ Wf32,
                       float* __restrict__ H, float* __restrict__ partial,
                       long long P, int ntiles) {
    __shared__ __align__(16) _Float16 wlds[128 * WSTR];
    __shared__ __align__(16) _Float16 ylds[2][16 * YSTR];

    int t = threadIdx.x;
    int lane = t & 31;
    int wave = t >> 5;                 // 0..7 -> o-tile
    for (int e = t; e < 128 * 128; e += GEMM_T) {
        int o = e >> 7, c = e & 127;
        wlds[o * WSTR + c] = (_Float16)Wf32[e];
    }

    int m    = lane & 15;
    int half = lane >> 4;
    int n    = lane & 15;
    int obase = wave * 16;

    int cpos = t >> 4;            // 0..15 position within tile (co-op staging)
    int cchk = (t & 15) * 8;      // 0..120 channel chunk (8 f16 = 16B)
    long long tstep = (long long)gridDim.x * 16;
    long long plim  = (long long)ntiles * 16;

    // per-thread LDS byte offsets (flat LDS aperture: low 32 bits = LDS offset)
    unsigned ldsoff[2];
    ldsoff[0] = (unsigned)(unsigned long long)(uintptr_t)&ylds[0][cpos * YSTR + cchk];
    ldsoff[1] = (unsigned)(unsigned long long)(uintptr_t)&ylds[1][cpos * YSTR + cchk];

    // prologue: async-stage first tile into buffer 0
    if (blockIdx.x < ntiles) {
        unsigned voff = (unsigned)((((long long)blockIdx.x * 16 + cpos) * 128 + cchk) * 2);
        asm volatile("global_load_async_to_lds_b128 %0, %1, %2"
                     :: "v"(ldsoff[0]), "v"(voff), "s"(Yact) : "memory");
    }

    __syncthreads();   // wlds ready

    // A fragments (weights, 16x32 f16 layout): two contiguous 8-f16 runs/chunk
    v16h afrag[4];
    {
        const _Float16* wrow = &wlds[(obase + m) * WSTR];
#pragma unroll
        for (int cc = 0; cc < 4; cc++) {
            v8h lo = *(const v8h*)&wrow[cc * 32 + half * 8];
            v8h hi = *(const v8h*)&wrow[cc * 32 + 16 + half * 8];
            v16h a;
#pragma unroll
            for (int j = 0; j < 8; j++) { a[j] = lo[j]; a[8 + j] = hi[j]; }
            afrag[cc] = a;
        }
    }

    float s[8], ss[8];
#pragma unroll
    for (int r = 0; r < 8; r++) { s[r] = 0.f; ss[r] = 0.f; }

    int bufi = 0;
    for (int tile = blockIdx.x; tile < ntiles; tile += gridDim.x) {
        long long p0 = (long long)tile * 16;
        // my async DMA for the current buffer has landed:
        asm volatile("s_wait_asynccnt 0x0" ::: "memory");
        // everyone's DMA visible; everyone done reading the other buffer:
        __syncthreads();
        // kick off DMA of the next tile into the alternate buffer
        long long pn = p0 + tstep;
        if (pn < plim) {
            unsigned voff = (unsigned)(((pn + cpos) * 128 + cchk) * 2);
            asm volatile("global_load_async_to_lds_b128 %0, %1, %2"
                         :: "v"(ldsoff[bufi ^ 1]), "v"(voff), "s"(Yact) : "memory");
        }

        const _Float16* yrow = &ylds[bufi][n * YSTR];
        // preload all 4 B fragments (8x ds_load_b128), then run the 4 WMMAs
        v16h bfr[4];
#pragma unroll
        for (int cc = 0; cc < 4; cc++) {
            v8h blo = *(const v8h*)&yrow[cc * 32 + half * 8];
            v8h bhi = *(const v8h*)&yrow[cc * 32 + 16 + half * 8];
            v16h bf;
#pragma unroll
            for (int j = 0; j < 8; j++) { bf[j] = blo[j]; bf[8 + j] = bhi[j]; }
            bfr[cc] = bf;
        }
        v8f acc = {};
#pragma unroll
        for (int cc = 0; cc < 4; cc++) {
            acc = __builtin_amdgcn_wmma_f32_16x16x32_f16(
                false, afrag[cc], false, bfr[cc], (short)0, acc, false, false);
        }
        // D rows o = obase + half*8 + r are contiguous -> one v8f store
        *(v8f*)&H[(p0 + n) * 128 + obase + half * 8] = acc;
#pragma unroll
        for (int r = 0; r < 8; r++) { float v = acc[r]; s[r] += v; ss[r] += v * v; }
        bufi ^= 1;
    }

    // reduce across each 16-lane half (wave32 shuffles)
#pragma unroll
    for (int r = 0; r < 8; r++) {
        for (int mask = 1; mask < 16; mask <<= 1) {
            s[r]  += __shfl_xor(s[r],  mask, 32);
            ss[r] += __shfl_xor(ss[r], mask, 32);
        }
    }
    if (n == 0) {
#pragma unroll
        for (int r = 0; r < 8; r++) {
            int o = obase + half * 8 + r;
            partial[((long long)blockIdx.x * 128 + o) * 2 + 0] = s[r];
            partial[((long long)blockIdx.x * 128 + o) * 2 + 1] = ss[r];
        }
    }
}

// ---------------------------------------------------------------------------
// Kernel 4: deterministic reduction of per-WG partials -> mean/var per channel
// ---------------------------------------------------------------------------
__global__ void k_bnstat(const float* __restrict__ partial, int nwg, float invcount,
                         float* __restrict__ stats) {
    int c = threadIdx.x;   // 128 threads
    float s = 0.f, ss = 0.f;
    for (int w = 0; w < nwg; w++) {
        s  += partial[((long long)w * 128 + c) * 2 + 0];
        ss += partial[((long long)w * 128 + c) * 2 + 1];
    }
    float mean = s * invcount;
    float var  = ss * invcount - mean * mean;
    stats[c * 2 + 0] = mean;
    stats[c * 2 + 1] = var;
}

// ---------------------------------------------------------------------------
// Kernel 5: BN apply + GELU (+ residual), position-major.
//   mode 0: gelu(bn(H)) -> f16     mode 1: gelu(bn(H)+res) -> f16 (in-place ok)
//   mode 2: gelu(bn(H)+res) -> f32 d_out remapped to (b,c,s)
// ---------------------------------------------------------------------------
__global__ void k_apply(const float* __restrict__ H, const float* __restrict__ stats,
                        const float* __restrict__ g, const float* __restrict__ bta,
                        const _Float16* __restrict__ res, _Float16* __restrict__ dstH,
                        float* __restrict__ dstF, long long P, int mode, int S2) {
    long long tot = P * 128;
    long long stride = (long long)gridDim.x * blockDim.x;
    for (long long i = (long long)blockIdx.x * blockDim.x + threadIdx.x; i < tot; i += stride) {
        int c = (int)(i & 127);
        long long p = i >> 7;
        float mean = stats[c * 2], var = stats[c * 2 + 1];
        float xv = g[c] * (H[i] - mean) * rsqrtf(var + BN_EPS) + bta[c];
        if (mode >= 1) xv += (float)res[i];
        float o = gelu_exact(xv);
        if (mode == 2) {
            long long b = p / S2, sp = p % S2;
            dstF[(b * 128 + c) * S2 + sp] = o;
        } else {
            dstH[i] = (_Float16)o;
        }
    }
}

// ---------------------------------------------------------------------------
// Kernel 6: max over K=32 -> pos-stage input Z[(b*1024+s)*128 + c]
// ---------------------------------------------------------------------------
__global__ void k_maxpool(const _Float16* __restrict__ A, _Float16* __restrict__ Z,
                          int M) {
    int i = blockIdx.x * blockDim.x + threadIdx.x;  // over M*128
    if (i >= M * 128) return;
    int c = i & 127, m = i >> 7;
    const _Float16* src = A + (long long)m * 32 * 128 + c;
    float mx = (float)src[0];
    for (int l = 1; l < 32; l++) { float v = (float)src[(long long)l * 128]; mx = v > mx ? v : mx; }
    Z[(long long)m * 128 + c] = (_Float16)mx;
}

// ---------------------------------------------------------------------------
extern "C" void kernel_launch(void* const* d_in, const int* in_sizes, int n_in,
                              void* d_out, int out_size, void* d_ws, size_t ws_size,
                              hipStream_t stream) {
    (void)in_sizes; (void)n_in; (void)out_size; (void)ws_size;
    const int B = 8, N = 4096, S = 1024;
    const long long P1 = (long long)B * S * 32;   // 262144 positions (pre stage)
    const long long P2 = (long long)B * S;        // 8192 positions (pos stage)
    const int NWG1 = 512, NWG2 = 128;

    const float*     xyz  = (const float*)d_in[0];
    const float*     x    = (const float*)d_in[1];
    const long long* fps  = (const long long*)d_in[2];
    const float* preW  = (const float*)d_in[3];
    const float* preG  = (const float*)d_in[4];
    const float* preB  = (const float*)d_in[5];
    const float* preFW = (const float*)d_in[6];
    const float* preFG = (const float*)d_in[7];
    const float* preFB = (const float*)d_in[8];
    const float* posW  = (const float*)d_in[9];
    const float* posG  = (const float*)d_in[10];
    const float* posB  = (const float*)d_in[11];
    const float* posFW = (const float*)d_in[12];
    const float* posFG = (const float*)d_in[13];
    const float* posFB = (const float*)d_in[14];
    float* out = (float*)d_out;

    // workspace carve-up (~272 MB)
    size_t off = 0;
    auto carve = [&](size_t bytes) {
        void* p = (char*)d_ws + off;
        off += (bytes + 255) & ~(size_t)255;
        return p;
    };
    _Float16* actA    = (_Float16*)carve(128 * P1 * 2);       // features / residual
    _Float16* actB    = (_Float16*)carve(128 * P1 * 2);       // intra-block temp
    float*    Hbuf    = (float*)   carve(128 * P1 * 4);       // conv output (f32)
    _Float16* Z       = (_Float16*)carve(128 * P2 * 2);       // pos-stage act
    _Float16* Ztmp    = (_Float16*)carve(128 * P2 * 2);
    float*    partial = (float*)   carve((size_t)NWG1 * 128 * 2 * 4);
    float*    stats   = (float*)   carve(128 * 2 * 4);

    auto run_conv = [&](const _Float16* in, const float* w, const float* g, const float* bb,
                        const _Float16* res, _Float16* outH, float* outF,
                        long long P, int nwg, int mode) {
        int ntiles = (int)(P / 16);
        k_gemm<<<nwg, GEMM_T, 0, stream>>>(in, w, Hbuf, partial, P, ntiles);
        k_bnstat<<<1, 128, 0, stream>>>(partial, nwg, 1.0f / (float)P, stats);
        k_apply<<<2048, 256, 0, stream>>>(Hbuf, stats, g, bb, res, outH, outF, P, mode, S);
    };

    // 1) new_xyz
    k_newxyz<<<(B * S + 255) / 256, 256, 0, stream>>>(xyz, fps, out, B, N, S);
    // 2) KNN + grouped feature build -> actA (f16, position-major)
    k_knn_feat<<<B * S, KNN_T, 0, stream>>>(xyz, x, fps, actA, B, N, S);

    // 3) pre-extraction: 2 res blocks + ffn  (P = 262144)
    for (int blk = 0; blk < 2; blk++) {
        run_conv(actA, preW + (size_t)(blk * 2 + 0) * 16384, preG + (blk * 2 + 0) * 128,
                 preB + (blk * 2 + 0) * 128, nullptr, actB, nullptr, P1, NWG1, 0);
        run_conv(actB, preW + (size_t)(blk * 2 + 1) * 16384, preG + (blk * 2 + 1) * 128,
                 preB + (blk * 2 + 1) * 128, actA, actA, nullptr, P1, NWG1, 1);
    }
    run_conv(actA, preFW, preFG, preFB, actA, actA, nullptr, P1, NWG1, 1);

    // 4) max-pool over K -> Z (position-major)
    k_maxpool<<<((int)P2 * 128 + 255) / 256, 256, 0, stream>>>(actA, Z, (int)P2);

    // 5) pos-extraction (P = 8192), final conv writes f32 into d_out (B,C,S)
    for (int blk = 0; blk < 2; blk++) {
        run_conv(Z, posW + (size_t)(blk * 2 + 0) * 16384, posG + (blk * 2 + 0) * 128,
                 posB + (blk * 2 + 0) * 128, nullptr, Ztmp, nullptr, P2, NWG2, 0);
        run_conv(Ztmp, posW + (size_t)(blk * 2 + 1) * 16384, posG + (blk * 2 + 1) * 128,
                 posB + (blk * 2 + 1) * 128, Z, Z, nullptr, P2, NWG2, 1);
    }
    run_conv(Z, posFW, posFG, posFB, Z, nullptr, out + (size_t)B * S * 3, P2, NWG2, 2);
}
